// OneFormerNeighborhoodAttention_17884243821355
// MI455X (gfx1250) — compile-verified, matrix operand order checked by reading, not run on hardware
//
#include <hip/hip_runtime.h>
#include <hip/hip_bf16.h>
#include <math.h>

// ---------------------------------------------------------------------------
// Problem constants (from reference): B=2, H=W=56, dim=256, heads=8, hd=32, k=7
// ---------------------------------------------------------------------------
#define BATCH   2
#define HH      56
#define WW      56
#define DIM     256
#define HEADS   8
#define HD      32
#define KWIN    7
#define NB      (KWIN * KWIN)          // 49 neighbors
#define M_TOT   (BATCH * HH * WW)      // 6272 pixels
#define QK_SCALE 0.17677669529663687f  // 1/sqrt(32)

typedef __attribute__((ext_vector_type(16))) __bf16          v16bf;
typedef __attribute__((ext_vector_type(16))) unsigned short  v16u16;
typedef __attribute__((ext_vector_type(8)))  unsigned short  v8u16;
typedef __attribute__((ext_vector_type(4)))  unsigned short  v4u16;
typedef __attribute__((ext_vector_type(8)))  float           v8f;

// float -> bf16 (round to nearest even)
static __device__ __forceinline__ unsigned short f2bf(float f) {
    unsigned u = __float_as_uint(f);
    u += 0x7FFFu + ((u >> 16) & 1u);
    return (unsigned short)(u >> 16);
}

// ---------------------------------------------------------------------------
// Kernel 0: bulk f32 -> bf16 conversion (grid-stride free; n % 4 == 0).
// ---------------------------------------------------------------------------
__global__ __launch_bounds__(256)
void cvt_f32_bf16(const float* __restrict__ src, unsigned short* __restrict__ dst,
                  int n) {
    const int i = (blockIdx.x * blockDim.x + threadIdx.x) * 4;
    if (i + 3 < n) {
        const float4 f = *(const float4*)(src + i);
        v4u16 o;
        o[0] = f2bf(f.x); o[1] = f2bf(f.y); o[2] = f2bf(f.z); o[3] = f2bf(f.w);
        *(v4u16*)(dst + i) = o;
    }
}

// Fragment layout (ISA 7.12.2, bf16 16x16x32):
//  lanes 0-15: VGPR 0-3 -> K=0..7, VGPR 4-7 -> K=16..23; lanes 16-31: +8.
//  => per lane: two contiguous 8-element (16 B) runs at (8*half) and (16+8*half).
static __device__ __forceinline__ v16bf load_frag(const unsigned short* row,
                                                  int kbase, int half) {
    const v8u16 lo = *(const v8u16*)(row + kbase + 8 * half);
    const v8u16 hi = *(const v8u16*)(row + kbase + 8 * half + 16);
    const v16u16 u = __builtin_shufflevector(lo, hi, 0, 1, 2, 3, 4, 5, 6, 7,
                                             8, 9, 10, 11, 12, 13, 14, 15);
    return __builtin_bit_cast(v16bf, u);
}

// ---------------------------------------------------------------------------
// Kernel 1: QKV projection GEMM on preconverted bf16 operands.
// y = x @ W^T + b  (Q additionally * 1/sqrt(hd)).
// One wave computes a 16(M) x 64(N) tile; blockIdx.z selects Q/K/V.
// Inner loop = b128 loads + v_wmma only (no conversion VALU).
// ---------------------------------------------------------------------------
__global__ __launch_bounds__(256)
void natten_qkv_wmma(const unsigned short* __restrict__ Xbf,
                     const unsigned short* __restrict__ Wbf,
                     const float* __restrict__ Bq, const float* __restrict__ Bk,
                     const float* __restrict__ Bv,
                     float* __restrict__ Qo, float* __restrict__ Ko,
                     float* __restrict__ Vo) {
    const int z = blockIdx.z;
    const unsigned short* Wm = Wbf + (size_t)z * DIM * DIM;
    const float* Bi = (z == 0) ? Bq : (z == 1) ? Bk : Bv;
    float*       Yo = (z == 0) ? Qo : (z == 1) ? Ko : Vo;

    const int lane  = threadIdx.x & 31;
    const int wid   = blockIdx.x * 8 + (threadIdx.x >> 5); // 1568 waves total
    const int mtile = wid >> 2;        // 0..391
    const int nt64  = wid & 3;         // 0..3  (64-wide N strip)
    const int m0    = mtile * 16;
    const int n0    = nt64 * 64;

    const int lrow  = lane & 15;       // row-in-tile for A, col-in-tile for B/D
    const int half  = lane >> 4;       // K half selector

    const unsigned short* Arow = Xbf + (size_t)(m0 + lrow) * DIM;
    const unsigned short* Brow0 = Wm + (size_t)(n0 + lrow) * DIM;

    v8f acc[4];
#pragma unroll
    for (int t = 0; t < 4; ++t) acc[t] = (v8f){0,0,0,0,0,0,0,0};

#pragma unroll
    for (int ks = 0; ks < DIM / 32; ++ks) {
        const int kbase = ks * 32;
        const v16bf av = load_frag(Arow, kbase, half);
#pragma unroll
        for (int t = 0; t < 4; ++t) {
            const v16bf bvf = load_frag(Brow0 + (size_t)t * 16 * DIM, kbase, half);
            acc[t] = __builtin_amdgcn_wmma_f32_16x16x32_bf16(
                false, av, false, bvf, (short)0, acc[t], false, false);
        }
    }

    // ---- epilogue: bias (+ Q scaling), store D ---------------------------
    const float scale = (z == 0) ? QK_SCALE : 1.0f;
#pragma unroll
    for (int t = 0; t < 4; ++t) {
        const int n = n0 + t * 16 + lrow;
        const float bias = Bi[n];
#pragma unroll
        for (int v = 0; v < 8; ++v) {
            const int m = m0 + v + 8 * half;
            Yo[(size_t)m * DIM + n] = (acc[t][v] + bias) * scale;
        }
    }
}

// window start: min(max(i-3,0), L-7)
static __device__ __forceinline__ int win_start(int i, int L) {
    int s = i - (KWIN / 2);
    s = s < 0 ? 0 : s;
    int mx = L - KWIN;
    return s > mx ? mx : s;
}

// ---------------------------------------------------------------------------
// Kernel 2: neighborhood attention.  One wave per (b, head, i, j).
// lane = head-dim element -> coalesced 128B loads of each K/V vector.
// ---------------------------------------------------------------------------
__global__ __launch_bounds__(256)
void natten_attn(const float* __restrict__ Q, const float* __restrict__ K,
                 const float* __restrict__ V, const float* __restrict__ rpb,
                 float* __restrict__ out) {
    const int lane = threadIdx.x & 31;
    const int wid  = blockIdx.x * 8 + (threadIdx.x >> 5);  // 50176 waves

    const int j    = wid % WW;
    const int i    = (wid / WW) % HH;
    const int head = (wid / (WW * HH)) % HEADS;
    const int b    = wid / (WW * HH * HEADS);

    const int is = win_start(i, HH);
    const int js = win_start(j, WW);

    const int pix = (b * HH + i) * WW + j;
    const int hoff = head * HD + lane;

    const float q = Q[(size_t)pix * DIM + hoff];   // already scaled by 1/sqrt(hd)
    const float* rp = rpb + head * (2 * KWIN - 1) * (2 * KWIN - 1);

    // ---- scores: 49 dot products with cross-lane tree reduction ----------
    float s0 = -1e30f, s1 = -1e30f, smax = -1e30f;
#pragma unroll
    for (int n = 0; n < NB; ++n) {
        const int di = n / KWIN, dj = n % KWIN;
        const int ki = is + di,  kj = js + dj;
        const int npix = (b * HH + ki) * WW + kj;
        float d = q * K[(size_t)npix * DIM + hoff];
#pragma unroll
        for (int off = 16; off > 0; off >>= 1)
            d += __shfl_xor(d, off, 32);
        d += rp[(ki - i + KWIN - 1) * (2 * KWIN - 1) + (kj - j + KWIN - 1)];
        smax = fmaxf(smax, d);
        if ((n & 31) == lane) { if (n < 32) s0 = d; else s1 = d; }
    }

    // ---- softmax ---------------------------------------------------------
    float p0 = __expf(s0 - smax);
    float p1 = __expf(s1 - smax);
    float ssum = p0 + p1;
#pragma unroll
    for (int off = 16; off > 0; off >>= 1)
        ssum += __shfl_xor(ssum, off, 32);
    const float inv = 1.0f / ssum;
    p0 *= inv;
    p1 *= inv;

    // ---- context: ctx[d] = sum_n p_n * V[n][d] ---------------------------
    float ctx = 0.0f;
#pragma unroll
    for (int n = 0; n < NB; ++n) {
        const float psrc = (n < 32) ? p0 : p1;
        const float pn = __shfl(psrc, n & 31, 32);
        const int di = n / KWIN, dj = n % KWIN;
        const int npix = (b * HH + (is + di)) * WW + (js + dj);
        ctx = fmaf(pn, V[(size_t)npix * DIM + hoff], ctx);
    }

    out[(size_t)pix * DIM + hoff] = ctx;
}

// ---------------------------------------------------------------------------
extern "C" void kernel_launch(void* const* d_in, const int* in_sizes, int n_in,
                              void* d_out, int out_size, void* d_ws, size_t ws_size,
                              hipStream_t stream) {
    const float* hidden = (const float*)d_in[0];
    const float* wq = (const float*)d_in[1];
    const float* bq = (const float*)d_in[2];
    const float* wk = (const float*)d_in[3];
    const float* bk = (const float*)d_in[4];
    const float* wv = (const float*)d_in[5];
    const float* bv = (const float*)d_in[6];
    const float* rpb = (const float*)d_in[7];
    float* out = (float*)d_out;

    const size_t E = (size_t)M_TOT * DIM;          // 1,605,632 elements
    float* Qb = (float*)d_ws;
    float* Kb = Qb + E;
    float* Vb = Kb + E;
    unsigned short* Xbf = (unsigned short*)(Vb + E);
    unsigned short* Wbf = Xbf + E;                 // 3 x 256 x 256 bf16

    // --- stage 0: precision conversion (once; W no longer re-converted 392x)
    cvt_f32_bf16<<<(int)(E / 4 + 255) / 256, 256, 0, stream>>>(hidden, Xbf, (int)E);
    const int wn = DIM * DIM;                      // 65,536
    cvt_f32_bf16<<<(wn / 4 + 255) / 256, 256, 0, stream>>>(wq, Wbf,          wn);
    cvt_f32_bf16<<<(wn / 4 + 255) / 256, 256, 0, stream>>>(wk, Wbf + wn,     wn);
    cvt_f32_bf16<<<(wn / 4 + 255) / 256, 256, 0, stream>>>(wv, Wbf + 2 * wn, wn);

    // --- stage 1: QKV GEMM. 392 M-tiles x 4 N64-strips = 1568 waves
    //     = 196 blocks of 8 waves; z selects Q / K / V.
    dim3 ggrid(196, 1, 3);
    natten_qkv_wmma<<<ggrid, 256, 0, stream>>>(Xbf, Wbf, bq, bk, bv, Qb, Kb, Vb);

    // --- stage 2: attention. 50,176 waves -> 6272 blocks of 8 waves.
    const int attn_blocks = (BATCH * HEADS * HH * WW) / 8;
    natten_attn<<<attn_blocks, 256, 0, stream>>>(Qb, Kb, Vb, rpb, out);
}
